// SimplexAttentionLayer_47837345743370
// MI455X (gfx1250) — matrix operand order repaired and checked by reading, not compiled, not run on hardware
//
#include <hip/hip_runtime.h>
#include <hip/hip_bf16.h>

typedef __bf16 bf16;
typedef __attribute__((ext_vector_type(16))) __bf16 v16bf;
typedef __attribute__((ext_vector_type(8)))  float  v8f;
typedef __attribute__((ext_vector_type(4)))  int    v4i;

#define HEADS 4
#define NN0   30000
#define NN1   90000
#define NN2   60000
#define DIN   256
#define DOUT  128
#define NNZ0  960000
#define NNZ12 180000
#define NTOT  (NN0 + NN1 + NN2)   // 180000

// ---- WMMA bf16 16x16x32 per-lane K order used here (contiguous; A and B agree) ----
// A (16x32, 16-bit): lane l holds row (l&15); half-elem i -> k = hi*16 + i
__device__ __forceinline__ int wka(int i, int hi) {
  return (hi << 4) + i;
}
// B (32x16, 16-bit): lane l holds col (l&15); half-elem i -> k = hi*16 + i
// (K-contiguous per lane, so B tiles live K-fastest in LDS)

__device__ __forceinline__ uint pack2(float lo, float hi) {
  union { uint u; bf16 h[2]; } p;
  p.h[0] = (bf16)lo; p.h[1] = (bf16)hi;
  return p.u;
}

// ---- async global->LDS 128-bit copy (GLOBAL_LOAD_ASYNC_TO_LDS_B128) ----
__device__ __forceinline__ void async_b128(const void* g, void* lds) {
#if __has_builtin(__builtin_amdgcn_global_load_async_to_lds_b128)
  __builtin_amdgcn_global_load_async_to_lds_b128(
      (__attribute__((address_space(1))) v4i*)(uintptr_t)g,
      (__attribute__((address_space(3))) v4i*)(unsigned)(uintptr_t)lds,
      0, 0);
#else
  unsigned l = (unsigned)(uintptr_t)lds;          // LDS byte offset
  unsigned long long ga = (unsigned long long)(uintptr_t)g;
  asm volatile("global_load_async_to_lds_b128 %0, %1, off"
               :: "v"(l), "v"(ga) : "memory");
#endif
}

__device__ __forceinline__ void wait_async0() {
#if __has_builtin(__builtin_amdgcn_s_wait_asynccnt)
  __builtin_amdgcn_s_wait_asynccnt(0);
#else
  asm volatile("s_wait_asynccnt 0" ::: "memory");
#endif
}

// ---- fp32 [H][K][N] -> bf16 transposed [H][N][K] (one-time, tiny) ----
__global__ void cvt_transpose(const float* __restrict__ in, bf16* __restrict__ out,
                              int K, int N, int n /* = H*K*N */) {
  int i = blockIdx.x * blockDim.x + threadIdx.x;
  if (i >= n) return;
  int h = i / (K * N);
  int r = (i / N) % K;     // k index
  int c = i % N;           // n index
  out[(size_t)h * N * K + (size_t)c * K + r] = (bf16)in[i];
}

// -------------------- Y = relu(X @ W + b), X fp32, Wt bf16[N,K], Y bf16 -----
__global__ __launch_bounds__(128)
void gemm_relu(const float* __restrict__ X, const bf16* __restrict__ Wt,
               const float* __restrict__ bias, bf16* __restrict__ Y,
               int M, int K) {
  __shared__ bf16 As[64][32];   // [m][k]
  __shared__ bf16 Bt[64][32];   // [n][k]  (K-fastest: per-lane fragments contiguous)
  const int m0 = blockIdx.x * 64;
  const int n0 = blockIdx.y * 64;
  const int t = threadIdx.x;
  const int wave = t >> 5, lane = t & 31;
  const int lm = lane & 15, hi = lane >> 4;

  v8f acc[4] = {};

  for (int k0 = 0; k0 < K; k0 += 32) {
    { // B tile: async DMA, overlapped with A-tile conversion below
      int row = t >> 1, kb = (t & 1) * 16;
      const bf16* srcp = Wt + (size_t)(n0 + row) * K + k0 + kb;
      async_b128(srcp,     &Bt[row][kb]);
      async_b128(srcp + 8, &Bt[row][kb + 8]);
    }
    { // A tile: 64x32 fp32 -> bf16, 16 elems/thread, pair-packed LDS stores
      int row = t >> 1, cb = (t & 1) * 16;
      int gm = m0 + row;
      const float* src = X + (size_t)gm * K + k0 + cb;
#pragma unroll
      for (int j = 0; j < 16; j += 4) {
        float4 v = (gm < M) ? *(const float4*)(src + j)
                            : make_float4(0.f, 0.f, 0.f, 0.f);
        *(uint*)&As[row][cb + j]     = pack2(v.x, v.y);
        *(uint*)&As[row][cb + j + 2] = pack2(v.z, v.w);
      }
    }
    wait_async0();
    __syncthreads();

    v16bf a;
#pragma unroll
    for (int i = 0; i < 16; ++i) a[i] = As[wave * 16 + lm][wka(i, hi)];
#pragma unroll
    for (int s = 0; s < 4; ++s) {
      v16bf b;
#pragma unroll
      for (int i = 0; i < 16; ++i) b[i] = Bt[s * 16 + lm][(hi << 4) + i];
      acc[s] = __builtin_amdgcn_wmma_f32_16x16x32_bf16(
          false, a, false, b, (short)0, acc[s], false, false);
    }
    __syncthreads();
  }

#pragma unroll
  for (int s = 0; s < 4; ++s) {
    int gn = n0 + s * 16 + lm;
    float bv = bias[gn];
#pragma unroll
    for (int i = 0; i < 8; ++i) {
      int gm = m0 + wave * 16 + (hi << 3) + i;
      if (gm < M) {
        float v = acc[s][i] + bv;
        Y[(size_t)gm * DOUT + gn] = (bf16)(v > 0.0f ? v : 0.0f);
      }
    }
  }
}

// ---- out += scale*(cat(xi0, agg0, agg1, agg2) @ Wagg + bagg), K=512 ----
// Waggt is transposed bf16 [N=128][K=512]
__global__ __launch_bounds__(128)
void gemm_out(const bf16* __restrict__ xi0, const float* __restrict__ agg,
              const bf16* __restrict__ Waggt, const float* __restrict__ bias,
              float* __restrict__ out, float scale) {
  __shared__ bf16 As[64][32];
  __shared__ bf16 Bt[64][32];
  const int m0 = blockIdx.x * 64;
  const int n0 = blockIdx.y * 64;
  const int t = threadIdx.x;
  const int wave = t >> 5, lane = t & 31;
  const int lm = lane & 15, hi = lane >> 4;

  v8f acc[4] = {};

  for (int k0 = 0; k0 < 4 * DOUT; k0 += 32) {
    const int src = k0 >> 7;     // 0: xi0 (bf16), 1..3: agg segments (f32)
    const int kin = k0 & 127;
    { // B tile: async DMA from transposed weights
      int row = t >> 1, kb = (t & 1) * 16;
      const bf16* srcw = Waggt + (size_t)(n0 + row) * (4 * DOUT) + k0 + kb;
      async_b128(srcw,     &Bt[row][kb]);
      async_b128(srcw + 8, &Bt[row][kb + 8]);
    }
    {
      int row = t >> 1, cb = (t & 1) * 16;
      int gm = m0 + row;
      if (src == 0) {
        const bf16* p = xi0 + (size_t)gm * DOUT + kin + cb;
        if (gm < NN0) {
          *(uint4*)&As[row][cb]     = *(const uint4*)p;
          *(uint4*)&As[row][cb + 8] = *(const uint4*)(p + 8);
        } else {
          uint4 z = make_uint4(0, 0, 0, 0);
          *(uint4*)&As[row][cb]     = z;
          *(uint4*)&As[row][cb + 8] = z;
        }
      } else {
        const float* p = agg + ((size_t)(src - 1) * NN0 + gm) * DOUT + kin + cb;
#pragma unroll
        for (int j = 0; j < 16; j += 4) {
          float4 v = (gm < NN0) ? *(const float4*)(p + j)
                                : make_float4(0.f, 0.f, 0.f, 0.f);
          *(uint*)&As[row][cb + j]     = pack2(v.x, v.y);
          *(uint*)&As[row][cb + j + 2] = pack2(v.z, v.w);
        }
      }
    }
    wait_async0();
    __syncthreads();

    v16bf a;
#pragma unroll
    for (int i = 0; i < 16; ++i) a[i] = As[wave * 16 + lm][wka(i, hi)];
#pragma unroll
    for (int s = 0; s < 4; ++s) {
      v16bf b;
#pragma unroll
      for (int i = 0; i < 16; ++i) b[i] = Bt[s * 16 + lm][(hi << 4) + i];
      acc[s] = __builtin_amdgcn_wmma_f32_16x16x32_bf16(
          false, a, false, b, (short)0, acc[s], false, false);
    }
    __syncthreads();
  }

#pragma unroll
  for (int s = 0; s < 4; ++s) {
    int gn = n0 + s * 16 + lm;
    float bv = bias[gn];
#pragma unroll
    for (int i = 0; i < 8; ++i) {
      int gm = m0 + wave * 16 + (hi << 3) + i;
      if (gm < NN0)
        out[(size_t)gm * DOUT + gn] += scale * (acc[s][i] + bv);
    }
  }
}

// -------- per-row attention score: o[r] = dot(Y[r,:], w) + b (wave/row) -----
__global__ __launch_bounds__(256)
void score(const bf16* __restrict__ Y, const float* __restrict__ w,
           const float* __restrict__ bscalar, float* __restrict__ o, int n) {
  int gid = blockIdx.x * blockDim.x + threadIdx.x;
  int wid = gid >> 5, lane = gid & 31;
  if (wid >= n) return;
  const bf16* row = Y + (size_t)wid * DOUT;
  float s = 0.0f;
#pragma unroll
  for (int d = 0; d < DOUT; d += 32) s += (float)row[d + lane] * w[d + lane];
#pragma unroll
  for (int off = 16; off > 0; off >>= 1) s += __shfl_xor(s, off, 32);
  if (lane == 0) o[wid] = s + bscalar[0];
}

// -------- attention scatter: agg[r,:] += sigmoid(a1[r]+a2[c]) * Y[c,:] ------
__global__ __launch_bounds__(256)
void scatter(const int* __restrict__ rows, const int* __restrict__ cols, int nnz,
             const float* __restrict__ a1, const float* __restrict__ a2, int a2off,
             const bf16* __restrict__ Y, int yoff, float* __restrict__ agg) {
  int gid = blockIdx.x * blockDim.x + threadIdx.x;
  int e = gid >> 5, lane = gid & 31;
  if (e >= nnz) return;
  int r = rows[e], c = cols[e];
  float z = a1[r] + a2[a2off + c];
  float att = 1.0f / (1.0f + __expf(-z));
  const bf16* srcp = Y + (size_t)(yoff + c) * DOUT;
  float* dst = agg + (size_t)r * DOUT;
#pragma unroll
  for (int d = 0; d < DOUT; d += 32)
    atomicAdd(dst + d + lane, att * (float)srcp[d + lane]);
}

extern "C" void kernel_launch(void* const* d_in, const int* in_sizes, int n_in,
                              void* d_out, int out_size, void* d_ws, size_t ws_size,
                              hipStream_t stream) {
  (void)in_sizes; (void)n_in; (void)out_size; (void)ws_size;

  const float* x0    = (const float*)d_in[0];
  const float* x1    = (const float*)d_in[1];
  const float* x2    = (const float*)d_in[2];
  const int*   rows0 = (const int*)d_in[3];
  const int*   cols0 = (const int*)d_in[4];
  const int*   rows1 = (const int*)d_in[5];
  const int*   cols1 = (const int*)d_in[6];
  const int*   rows2 = (const int*)d_in[7];
  const int*   cols2 = (const int*)d_in[8];
  const float* W1    = (const float*)d_in[9];
  const float* b1    = (const float*)d_in[10];
  const float* a1w   = (const float*)d_in[11];
  const float* a1b   = (const float*)d_in[12];
  const float* a2w   = (const float*)d_in[13];
  const float* a2b   = (const float*)d_in[14];
  const float* Wagg  = (const float*)d_in[15];
  const float* bagg  = (const float*)d_in[16];
  float* out = (float*)d_out;

  // ---- workspace carve-out (~94 MB) ----
  char* ws = (char*)d_ws;
  size_t off = 0;
  bf16*  W1t   = (bf16*)(ws + off); off += (size_t)HEADS * DIN * DOUT * 2;      // [H][128][256]
  bf16*  Waggt = (bf16*)(ws + off); off += (size_t)HEADS * 4 * DOUT * DOUT * 2; // [H][128][512]
  bf16*  Y     = (bf16*)(ws + off); off += (size_t)NTOT * DOUT * 2;             // per-head activations
  float* a1    = (float*)(ws + off); off += (size_t)NN0 * 4;
  float* a2    = (float*)(ws + off); off += (size_t)NTOT * 4;
  float* agg   = (float*)(ws + off); off += (size_t)3 * NN0 * DOUT * 4;         // per-head aggregations

  // one-time weight conversion + transpose to bf16 [N][K]
  { int n = HEADS * DIN * DOUT;
    cvt_transpose<<<(n + 255) / 256, 256, 0, stream>>>(W1, W1t, DIN, DOUT, n); }
  { int n = HEADS * 4 * DOUT * DOUT;
    cvt_transpose<<<(n + 255) / 256, 256, 0, stream>>>(Wagg, Waggt, 4 * DOUT, DOUT, n); }
  (void)hipMemsetAsync(out, 0, (size_t)NN0 * DOUT * sizeof(float), stream);

  dim3 blk(128);
  for (int h = 0; h < HEADS; ++h) {
    const bf16* W1h = W1t + (size_t)h * DIN * DOUT;

    // relu(X @ W1 + b1) for all three simplex levels, packed into Y (bf16)
    gemm_relu<<<dim3((NN0 + 63) / 64, 2), blk, 0, stream>>>(
        x0, W1h, b1 + h * DOUT, Y, NN0, DIN);
    gemm_relu<<<dim3((NN1 + 63) / 64, 2), blk, 0, stream>>>(
        x1, W1h, b1 + h * DOUT, Y + (size_t)NN0 * DOUT, NN1, DIN);
    gemm_relu<<<dim3((NN2 + 63) / 64, 2), blk, 0, stream>>>(
        x2, W1h, b1 + h * DOUT, Y + (size_t)(NN0 + NN1) * DOUT, NN2, DIN);

    // attention source/target scores
    score<<<(NN0 * 32 + 255) / 256, 256, 0, stream>>>(Y, a1w + h * DOUT, a1b + h, a1, NN0);
    score<<<(NTOT * 32 + 255) / 256, 256, 0, stream>>>(Y, a2w + h * DOUT, a2b + h, a2, NTOT);

    (void)hipMemsetAsync(agg, 0, (size_t)3 * NN0 * DOUT * sizeof(float), stream);

    // sparse attention aggregation (wave per nnz, L2-resident gather + atomics)
    scatter<<<(NNZ0 * 32 + 255) / 256, 256, 0, stream>>>(
        rows0, cols0, NNZ0, a1, a2, 0, Y, 0, agg);
    scatter<<<(NNZ12 * 32 + 255) / 256, 256, 0, stream>>>(
        rows1, cols1, NNZ12, a1, a2, NN0, Y, NN0, agg + (size_t)NN0 * DOUT);
    scatter<<<(NNZ12 * 32 + 255) / 256, 256, 0, stream>>>(
        rows2, cols2, NNZ12, a1, a2, NN0 + NN1, Y, NN0 + NN1, agg + (size_t)2 * NN0 * DOUT);

    // out += 0.25 * (cat @ Wagg_h + bagg_h)
    gemm_out<<<dim3((NN0 + 63) / 64, 2), blk, 0, stream>>>(
        Y, agg, Waggt + (size_t)h * 4 * DOUT * DOUT, bagg + h * DOUT, out, 1.0f / HEADS);
  }
}
